// TriOut_20194936225858
// MI455X (gfx1250) — matrix operand order repaired and checked by reading, not compiled, not run on hardware
//
#include <hip/hip_runtime.h>
#include <hip/hip_bf16.h>

// ---------------------------------------------------------------------------
// Triangular multiplicative update (outgoing) for MI455X / gfx1250.
// All GEMMs routed through v_wmma_f32_16x16x32_bf16 (bf16 in, fp32 accum).
// Kernel 2 stages its K-tiles with GLOBAL_LOAD_ASYNC_TO_LDS_B128 (ASYNCcnt),
// bypassing the register file and overlapping the next stage with WMMA work.
// ---------------------------------------------------------------------------

typedef __attribute__((ext_vector_type(8)))  __bf16 v8bf;
typedef __attribute__((ext_vector_type(16))) __bf16 v16bf;
typedef __attribute__((ext_vector_type(8)))  float  v8f;

#define LL   512
#define ZC   128          // z_dim == c == 128
#define NPOS (LL * LL)    // 262144

__device__ __forceinline__ unsigned short f2bf(float f) {
    unsigned int u = __float_as_uint(f);
    unsigned int r = u + 0x7FFFu + ((u >> 16) & 1u);   // round-to-nearest-even
    return (unsigned short)(r >> 16);
}
__device__ __forceinline__ float bf2f(unsigned short b) {
    return __uint_as_float(((unsigned int)b) << 16);
}
__device__ __forceinline__ float sigmoidf(float x) {
    return 1.0f / (1.0f + __expf(-x));
}

// A-fragment (M x K = 16 x 32 bf16): lane<16 -> row=lane, K {kk..kk+7, kk+16..kk+23}
//                                    lane>=16 -> row=lane-16, K {kk+8..15, kk+24..31}
__device__ __forceinline__ v16bf ldfragA(const unsigned short* buf, int stride,
                                         int row0, int kk, int lrow, int lhi) {
    const unsigned short* p = buf + (row0 + lrow) * stride + kk + lhi * 8;
    v8bf c0 = *(const v8bf*)(p);
    v8bf c1 = *(const v8bf*)(p + 16);
    return __builtin_shufflevector(c0, c1, 0,1,2,3,4,5,6,7,8,9,10,11,12,13,14,15);
}
// B-fragment (K x N = 32 x 16 bf16), stored as rows of B^T (row = n, k contiguous):
// lane<16 -> col=lane, K {kk..kk+7, kk+16..kk+23}; lane>=16 -> col=lane-16, K {kk+8..15, kk+24..31}
__device__ __forceinline__ v16bf ldfragB(const unsigned short* buf, int stride,
                                         int row0, int kk, int lrow, int lhi) {
    const unsigned short* p = buf + (row0 + lrow) * stride + kk + lhi * 8;
    v8bf c0 = *(const v8bf*)(p);
    v8bf c1 = *(const v8bf*)(p + 16);
    return __builtin_shufflevector(c0, c1, 0,1,2,3,4,5,6,7,8,9,10,11,12,13,14,15);
}
__device__ __forceinline__ v8f wmma_bf16(v16bf a, v16bf b, v8f c) {
    return __builtin_amdgcn_wmma_f32_16x16x32_bf16(false, a, false, b, (short)0, c,
                                                   false, false);
}
__device__ __forceinline__ v8f vzero8() {
    v8f v = {0.f, 0.f, 0.f, 0.f, 0.f, 0.f, 0.f, 0.f};
    return v;
}

// Async 16B global -> LDS copy (per enabled lane). LDS address is the low 32
// bits of the generic pointer (flat->LDS truncation per ISA 10.2); hardware
// adds LDS_BASE. Tracked by ASYNCcnt.
__device__ __forceinline__ void async_g2l_b128(void* lds_ptr, const void* gptr) {
    unsigned lds_a = (unsigned)(unsigned long long)lds_ptr;
    asm volatile("global_load_async_to_lds_b128 %0, %1, off"
                 :: "v"(lds_a), "v"(gptr) : "memory");
}
__device__ __forceinline__ void wait_async0() {
    asm volatile("s_wait_asynccnt 0x0" ::: "memory");
}

// ---------------------------------------------------------------------------
// Kernel 1: LayerNorm(z) + gated projection a = (zn@a_w^T+a_b)*sig(zn@ag_w^T+ag_b)
//           + gate = sig(zn@g_w^T+g_b).
// a written channel-major  a_t[c][pos]  (pos = i*512+l)  -> batched GEMM layout.
// gate written position-major gate[pos][z].
// One block = 128 consecutive positions (stays within one i-row), 256 threads.
// ---------------------------------------------------------------------------
__global__ __launch_bounds__(256) void triout_k1(
    const float* __restrict__ z, const float* __restrict__ ng, const float* __restrict__ nb,
    const float* __restrict__ a_w, const float* __restrict__ a_b,
    const float* __restrict__ ag_w, const float* __restrict__ ag_b,
    const float* __restrict__ g_w, const float* __restrict__ g_b,
    unsigned short* __restrict__ a_t, unsigned short* __restrict__ gate)
{
    __shared__ __align__(16) unsigned short znb[128 * 136];  // LN(z) bf16
    __shared__ __align__(16) unsigned short wA [128 * 136];  // a_w  (later g_w)
    __shared__ __align__(16) unsigned short wG [128 * 136];  // ag_w
    __shared__ __align__(16) unsigned short sst[128 * 136];  // staging for transpose

    const int tid  = threadIdx.x;
    const int base = blockIdx.x * 128;                       // position base

    // ---- LayerNorm: 2 threads per position, 64 channels each ----
    {
        const int p = tid >> 1, h = tid & 1;
        const float* zr = z + (size_t)(base + p) * ZC + h * 64;
        float v[64];
        float s = 0.f, s2 = 0.f;
#pragma unroll
        for (int j = 0; j < 64; j++) { float x = zr[j]; v[j] = x; s += x; s2 += x * x; }
        s  += __shfl_xor(s, 1, 32);
        s2 += __shfl_xor(s2, 1, 32);
        const float mean = s * (1.f / 128.f);
        const float var  = s2 * (1.f / 128.f) - mean * mean;
        const float rstd = rsqrtf(var + 1e-5f);
#pragma unroll
        for (int j = 0; j < 64; j++) {
            const int c = h * 64 + j;
            znb[p * 136 + c] = f2bf((v[j] - mean) * rstd * ng[c] + nb[c]);
        }
    }
    // ---- weights -> LDS bf16 ----
    for (int i = tid; i < ZC * ZC; i += 256) {
        wA[(i >> 7) * 136 + (i & 127)] = f2bf(a_w[i]);
        wG[(i >> 7) * 136 + (i & 127)] = f2bf(ag_w[i]);
    }
    __syncthreads();

    const int w = tid >> 5, lane = tid & 31;
    const int lrow = lane & 15, lhi = lane >> 4;
    const int pm = w;                                        // wave = one 16-pos tile

    // ---- a / ag GEMM (128x128x128), fused sigmoid gating ----
    {
        v16bf af[4];
#pragma unroll
        for (int j = 0; j < 4; j++) af[j] = ldfragA(znb, 136, pm * 16, j * 32, lrow, lhi);
        for (int cn = 0; cn < 8; cn++) {
            v8f accA = vzero8(), accG = vzero8();
#pragma unroll
            for (int j = 0; j < 4; j++) {
                v16bf ba = ldfragB(wA, 136, cn * 16, j * 32, lrow, lhi);
                v16bf bg = ldfragB(wG, 136, cn * 16, j * 32, lrow, lhi);
                accA = wmma_bf16(af[j], ba, accA);
                accG = wmma_bf16(af[j], bg, accG);
            }
            const int ch = cn * 16 + lrow;
            const float bA = a_b[ch], bG = ag_b[ch];
#pragma unroll
            for (int v = 0; v < 8; v++) {
                const float aval = (accA[v] + bA) * sigmoidf(accG[v] + bG);
                const int p = pm * 16 + v + lhi * 8;
                sst[ch * 136 + p] = f2bf(aval);              // transposed staging
            }
        }
    }
    __syncthreads();
    // ---- write a_t channel-major (contiguous 256B rows) + reload g_w ----
    for (int i = tid; i < ZC * 128; i += 256) {
        const int ch = i >> 7, p = i & 127;
        a_t[(size_t)ch * NPOS + base + p] = sst[ch * 136 + p];
    }
    for (int i = tid; i < ZC * ZC; i += 256)
        wA[(i >> 7) * 136 + (i & 127)] = f2bf(g_w[i]);
    __syncthreads();

    // ---- gate GEMM ----
    {
        v16bf af[4];
#pragma unroll
        for (int j = 0; j < 4; j++) af[j] = ldfragA(znb, 136, pm * 16, j * 32, lrow, lhi);
        for (int cn = 0; cn < 8; cn++) {
            v8f acc = vzero8();
#pragma unroll
            for (int j = 0; j < 4; j++) {
                v16bf bg = ldfragB(wA, 136, cn * 16, j * 32, lrow, lhi);
                acc = wmma_bf16(af[j], bg, acc);
            }
            const int ch = cn * 16 + lrow;
            const float bG = g_b[ch];
#pragma unroll
            for (int v = 0; v < 8; v++) {
                const int p = pm * 16 + v + lhi * 8;
                sst[p * 136 + ch] = f2bf(sigmoidf(acc[v] + bG));
            }
        }
    }
    __syncthreads();
    for (int i = tid; i < ZC * 128; i += 256) {
        const int p = i >> 7, ch = i & 127;
        gate[(size_t)(base + p) * ZC + ch] = sst[p * 136 + ch];
    }
}

// ---------------------------------------------------------------------------
// Kernel 2: o_t[c] = A_c @ A_c^T for A_c = a_t[c] (512x512 bf16), fp32 out.
// Upper-triangle 128x128 macrotiles (10 pairs) x 128 channels; symmetric
// mirror writes give the lower triangle (bitwise-identical values).
// 8 waves: wave = (wm in 0..3, wn in 0..1) -> 2x4 tiles of 16x16 each.
// K-tile staging via async global->LDS DMA, double buffered.
// ---------------------------------------------------------------------------
__global__ __launch_bounds__(256) void triout_k2(
    const unsigned short* __restrict__ a_t, float* __restrict__ o_t)
{
    const int c = blockIdx.y;
    int r = blockIdx.x, bi = 0, rowlen = 4;
    while (r >= rowlen) { r -= rowlen; rowlen--; bi++; }
    const int bj = bi + r;

    const unsigned short* __restrict__ A = a_t + (size_t)c * NPOS;
    float* __restrict__ O = o_t + (size_t)c * NPOS;
    const int ibase = bi * 128, jbase = bj * 128;

    __shared__ __align__(16) unsigned short Ab[2][128 * 40];
    __shared__ __align__(16) unsigned short Bb[2][128 * 40];

    const int tid = threadIdx.x, w = tid >> 5, lane = tid & 31;
    const int lrow = lane & 15, lhi = lane >> 4;
    const int wm = w & 3, wn = w >> 2;

    v8f acc[2][4];
#pragma unroll
    for (int mt = 0; mt < 2; mt++)
#pragma unroll
        for (int nt = 0; nt < 4; nt++) acc[mt][nt] = vzero8();

    // each thread DMAs 2x16B for A and 2x16B for B per stage (128 rows x 32 k)
    auto loadtile = [&](int buf, int kk) {
#pragma unroll
        for (int t0 = 0; t0 < 512; t0 += 256) {
            const int t = t0 + tid;
            const int row = t >> 2, seg = (t & 3) * 8;
            async_g2l_b128(&Ab[buf][row * 40 + seg], &A[(ibase + row) * LL + kk + seg]);
            async_g2l_b128(&Bb[buf][row * 40 + seg], &A[(jbase + row) * LL + kk + seg]);
        }
    };

    loadtile(0, 0);
    wait_async0();
    __syncthreads();
    for (int ks = 0; ks < 16; ks++) {
        if (ks < 15) loadtile((ks + 1) & 1, (ks + 1) * 32);   // overlap with WMMAs below
        const unsigned short* ab = Ab[ks & 1];
        const unsigned short* bb = Bb[ks & 1];
        v16bf af[2], bf[4];
#pragma unroll
        for (int mt = 0; mt < 2; mt++) af[mt] = ldfragA(ab, 40, wm * 32 + mt * 16, 0, lrow, lhi);
#pragma unroll
        for (int nt = 0; nt < 4; nt++) bf[nt] = ldfragB(bb, 40, wn * 64 + nt * 16, 0, lrow, lhi);
#pragma unroll
        for (int mt = 0; mt < 2; mt++)
#pragma unroll
            for (int nt = 0; nt < 4; nt++)
                acc[mt][nt] = wmma_bf16(af[mt], bf[nt], acc[mt][nt]);
        wait_async0();          // this wave's DMAs for the next stage have landed
        __syncthreads();        // everyone's DMAs landed + buffer ks&1 fully consumed
    }

#pragma unroll
    for (int mt = 0; mt < 2; mt++) {
#pragma unroll
        for (int nt = 0; nt < 4; nt++) {
            const int gj = jbase + wn * 64 + nt * 16 + lrow;
#pragma unroll
            for (int v = 0; v < 8; v++) {
                const int gi = ibase + wm * 32 + mt * 16 + v + lhi * 8;
                const float val = acc[mt][nt][v];
                O[(size_t)gi * LL + gj] = val;
                if (bi != bj) O[(size_t)gj * LL + gi] = val;   // symmetry
            }
        }
    }
}

// ---------------------------------------------------------------------------
// Kernel 3: LN over c, out = (LN(o) @ o_w^T + o_b) * gate.  128 positions/block.
// ---------------------------------------------------------------------------
__global__ __launch_bounds__(256) void triout_k3(
    const float* __restrict__ o_t,
    const float* __restrict__ on_g, const float* __restrict__ on_b,
    const float* __restrict__ o_w, const float* __restrict__ o_b,
    const unsigned short* __restrict__ gate, float* __restrict__ out)
{
    __shared__ __align__(16) float          obuf[128 * 129]; // gather + output stage
    __shared__ __align__(16) unsigned short onb [128 * 136]; // LN(o) bf16
    __shared__ __align__(16) unsigned short wo  [128 * 136]; // o_w bf16

    const int tid  = threadIdx.x;
    const int base = blockIdx.x * 128;

    // gather channel-major o_t into [pos][c] (contiguous 512B per channel row)
    for (int i = tid; i < ZC * 128; i += 256) {
        const int c = i >> 7, p = i & 127;
        obuf[p * 129 + c] = o_t[(size_t)c * NPOS + base + p];
    }
    for (int i = tid; i < ZC * ZC; i += 256)
        wo[(i >> 7) * 136 + (i & 127)] = f2bf(o_w[i]);
    __syncthreads();

    // ---- LayerNorm over c ----
    {
        const int p = tid >> 1, h = tid & 1;
        float v[64];
        float s = 0.f, s2 = 0.f;
#pragma unroll
        for (int j = 0; j < 64; j++) {
            float x = obuf[p * 129 + h * 64 + j];
            v[j] = x; s += x; s2 += x * x;
        }
        s  += __shfl_xor(s, 1, 32);
        s2 += __shfl_xor(s2, 1, 32);
        const float mean = s * (1.f / 128.f);
        const float var  = s2 * (1.f / 128.f) - mean * mean;
        const float rstd = rsqrtf(var + 1e-5f);
#pragma unroll
        for (int j = 0; j < 64; j++) {
            const int c = h * 64 + j;
            onb[p * 136 + c] = f2bf((v[j] - mean) * rstd * on_g[c] + on_b[c]);
        }
    }
    __syncthreads();

    // ---- output projection GEMM (128x128x128), bias fused; stage in obuf ----
    const int w = tid >> 5, lane = tid & 31;
    const int lrow = lane & 15, lhi = lane >> 4;
    const int pm = w;
    {
        v16bf af[4];
#pragma unroll
        for (int j = 0; j < 4; j++) af[j] = ldfragA(onb, 136, pm * 16, j * 32, lrow, lhi);
        for (int zn = 0; zn < 8; zn++) {
            v8f acc = vzero8();
#pragma unroll
            for (int j = 0; j < 4; j++) {
                v16bf bw = ldfragB(wo, 136, zn * 16, j * 32, lrow, lhi);
                acc = wmma_bf16(af[j], bw, acc);
            }
            const int zc = zn * 16 + lrow;
            const float bo = o_b[zc];
#pragma unroll
            for (int v = 0; v < 8; v++) {
                const int p = pm * 16 + v + lhi * 8;
                obuf[p * 129 + zc] = acc[v] + bo;
            }
        }
    }
    __syncthreads();

    // ---- multiply by gate, write fp32 output ----
    for (int i = tid; i < ZC * 128; i += 256) {
        const int p = i >> 7, zc = i & 127;
        const float g = bf2f(gate[(size_t)(base + p) * ZC + zc]);
        out[(size_t)(base + p) * ZC + zc] = obuf[p * 129 + zc] * g;
    }
}

// ---------------------------------------------------------------------------
extern "C" void kernel_launch(void* const* d_in, const int* in_sizes, int n_in,
                              void* d_out, int out_size, void* d_ws, size_t ws_size,
                              hipStream_t stream) {
    const float* z      = (const float*)d_in[0];
    const float* norm_g = (const float*)d_in[1];
    const float* norm_b = (const float*)d_in[2];
    const float* a_w    = (const float*)d_in[3];
    const float* a_b    = (const float*)d_in[4];
    const float* ag_w   = (const float*)d_in[5];
    const float* ag_b   = (const float*)d_in[6];
    const float* on_g   = (const float*)d_in[7];
    const float* on_b   = (const float*)d_in[8];
    const float* o_w    = (const float*)d_in[9];
    const float* o_b    = (const float*)d_in[10];
    const float* g_w    = (const float*)d_in[11];
    const float* g_b    = (const float*)d_in[12];
    float* out = (float*)d_out;

    // workspace: a_t bf16 (64MB) | gate bf16 (64MB) | o_t fp32 (128MB)
    unsigned short* a_t  = (unsigned short*)d_ws;
    unsigned short* gate = a_t + (size_t)NPOS * ZC;
    float*          o_t  = (float*)(gate + (size_t)NPOS * ZC);

    triout_k1<<<NPOS / 128, 256, 0, stream>>>(z, norm_g, norm_b, a_w, a_b,
                                              ag_w, ag_b, g_w, g_b, a_t, gate);
    triout_k2<<<dim3(10, ZC), 256, 0, stream>>>(a_t, o_t);
    triout_k3<<<NPOS / 128, 256, 0, stream>>>(o_t, on_g, on_b, o_w, o_b, gate, out);
}
